// MSA_64106681860467
// MI455X (gfx1250) — matrix operand order repaired and checked by reading, not compiled, not run on hardware
//
#include <hip/hip_runtime.h>

#define B_  4
#define H_  12
#define N_  2048
#define DH_ 64
#define D_  768

typedef __attribute__((ext_vector_type(16))) __bf16          v16bf;
typedef __attribute__((ext_vector_type(8)))  float           v8f;
typedef __attribute__((ext_vector_type(16))) unsigned short  usv16;
typedef __attribute__((ext_vector_type(8)))  unsigned short  usv8;

__device__ __forceinline__ unsigned short f2bf(float f) {
  unsigned int u = __builtin_bit_cast(unsigned int, f);
  u += 0x7fffu + ((u >> 16) & 1u);           // round-to-nearest
  return (unsigned short)(u >> 16);
}
__device__ __forceinline__ v16bf bfcast(usv16 u) { return __builtin_bit_cast(v16bf, u); }
__device__ __forceinline__ usv16 cat8(usv8 lo, usv8 hi) {
  return __builtin_shufflevector(lo, hi, 0,1,2,3,4,5,6,7,8,9,10,11,12,13,14,15);
}

// ---------------------------------------------------------------- fp32 -> bf16
__global__ void __launch_bounds__(256) cvt_bf16(const float* __restrict__ s,
                                                unsigned short* __restrict__ d, int n8) {
  int i = blockIdx.x * 256 + threadIdx.x;
  if (i >= n8) return;
  const float4* sp = (const float4*)s;
  float4 a = sp[2 * i], b = sp[2 * i + 1];
  usv8 o;
  o[0]=f2bf(a.x); o[1]=f2bf(a.y); o[2]=f2bf(a.z); o[3]=f2bf(a.w);
  o[4]=f2bf(b.x); o[5]=f2bf(b.y); o[6]=f2bf(b.z); o[7]=f2bf(b.w);
  ((usv8*)d)[i] = o;
}

// ------------------------------------------------- QKV projection: y = z @ W^T
// Wave tile: 32 rows x 64 cols (2 A-frags share 4 B-frags), K-step 32.
// A layout (16x32 bf16): lanes 0-15 row=ln hold K {0..7,16..23};
//   lanes 16-31 hold {8..15,24..31}  => two 16B chunks at off0, off0+16 (off0=0/8).
// B layout (32x16): lane col n=ln, same K order as A => two 16B chunks at off0, off0+16.
__global__ void __launch_bounds__(256) qkv_gemm(
    const unsigned short* __restrict__ zb, const unsigned short* __restrict__ Wb,
    unsigned short* __restrict__ Qb, unsigned short* __restrict__ Kb,
    unsigned short* __restrict__ Vt) {
  const int lane  = threadIdx.x & 31;
  const int wave  = threadIdx.x >> 5;
  const int wg    = blockIdx.x * 8 + wave;       // 256*36 wave tiles
  const int mtile = wg / 36;
  const int ctile = wg % 36;
  const int row0  = mtile * 32;
  const int c0    = ctile * 64;                  // one (head, q/k/v) slab
  const int h     = c0 / 192;
  const int s     = (c0 % 192) / 64;             // 0=Q 1=K 2=V
  const int ln    = lane & 15;
  const int off0  = (lane < 16) ? 0 : 8;
  const int boff  = (lane < 16) ? 0 : 16;

  const unsigned short* za0 = zb + (size_t)(row0 + ln) * D_;
  const unsigned short* za1 = zb + (size_t)(row0 + 16 + ln) * D_;
  v8f acc[2][4] = {};
  for (int kb = 0; kb < D_; kb += 32) {
    usv8 a0lo = *(const usv8*)(za0 + kb + off0);
    usv8 a0hi = *(const usv8*)(za0 + kb + off0 + 16);
    usv8 a1lo = *(const usv8*)(za1 + kb + off0);
    usv8 a1hi = *(const usv8*)(za1 + kb + off0 + 16);
    v16bf A0 = bfcast(cat8(a0lo, a0hi));
    v16bf A1 = bfcast(cat8(a1lo, a1hi));
#pragma unroll
    for (int sub = 0; sub < 4; ++sub) {
      const unsigned short* wr = Wb + (size_t)(c0 + sub * 16 + ln) * D_ + kb + off0;
      v16bf Bm = bfcast(cat8(*(const usv8*)wr, *(const usv8*)(wr + 16)));
      acc[0][sub] = __builtin_amdgcn_wmma_f32_16x16x32_bf16(
          false, A0, false, Bm, (short)0, acc[0][sub], false, false);
      acc[1][sub] = __builtin_amdgcn_wmma_f32_16x16x32_bf16(
          false, A1, false, Bm, (short)0, acc[1][sub], false, false);
    }
  }
  // C layout: element r -> row M = (lane<16 ? r : 8+r), col = sub*16 + ln
  const int bh = (row0 >> 11) * H_ + h;
#pragma unroll
  for (int g = 0; g < 2; ++g) {
    const int nb = (row0 & (N_ - 1)) + g * 16 + ((lane < 16) ? 0 : 8);
    if (s == 2) {                                 // V stored transposed [dh][n]
#pragma unroll
      for (int sub = 0; sub < 4; ++sub) {
        const int dh = sub * 16 + ln;
        usv8 o;
#pragma unroll
        for (int r = 0; r < 8; ++r) o[r] = f2bf(acc[g][sub][r]);
        *(usv8*)(Vt + ((size_t)bh * DH_ + dh) * N_ + nb) = o;
      }
    } else {
      unsigned short* dst = (s == 0 ? Qb : Kb) + (size_t)bh * N_ * DH_;
      const float scl = (s == 0) ? 0.125f : 1.0f; // fold 1/sqrt(64) into Q
#pragma unroll
      for (int sub = 0; sub < 4; ++sub) {
        const int dh = sub * 16 + ln;
#pragma unroll
        for (int r = 0; r < 8; ++r)
          dst[(size_t)(nb + r) * DH_ + dh] = f2bf(acc[g][sub][r] * scl);
      }
    }
  }
}

// ----------------------------------------------------- flash attention (bf16)
// Block: 8 waves, 128 q-rows of one (b,h). Wave: 16 q-rows, full DH=64 output.
__global__ void __launch_bounds__(256) attn(
    const unsigned short* __restrict__ Qb, const unsigned short* __restrict__ Kb,
    const unsigned short* __restrict__ Vt, float* __restrict__ out) {
  __shared__ __align__(32) unsigned short Klds[32 * 64];   // [key][dh]
  __shared__ __align__(32) unsigned short Vlds[64 * 32];   // [dh][key]
  __shared__ __align__(32) unsigned short Plds[8][16 * 32];// per-wave P tile

  const int tid  = threadIdx.x;
  const int lane = tid & 31;
  const int wave = tid >> 5;
  const int qblk = blockIdx.x & 15;
  const int bh   = blockIdx.x >> 4;
  const int bi   = bh / H_;
  const int h    = bh % H_;
  const int ln   = lane & 15;
  const int off0 = (lane < 16) ? 0 : 8;
  const int boff = (lane < 16) ? 0 : 16;
  const int prow = (lane < 16) ? 0 : 8;
  const int q0   = qblk * 128 + wave * 16;

  const unsigned short* Qh = Qb + (size_t)bh * N_ * DH_;
  const unsigned short* Kh = Kb + (size_t)bh * N_ * DH_;
  const unsigned short* Vh = Vt + (size_t)bh * DH_ * N_;

  // Q fragments for d-chunks 0..31 and 32..63 (loaded once)
  const unsigned short* qrow = Qh + (size_t)(q0 + ln) * DH_;
  v16bf aQ[2];
#pragma unroll
  for (int c = 0; c < 2; ++c) {
    usv8 lo = *(const usv8*)(qrow + c * 32 + off0);
    usv8 hi = *(const usv8*)(qrow + c * 32 + off0 + 16);
    aQ[c] = bfcast(cat8(lo, hi));
  }

  v8f o[4] = {};
  float mrow[8], lrow[8];
#pragma unroll
  for (int r = 0; r < 8; ++r) { mrow[r] = -3.0e38f; lrow[r] = 0.0f; }

  for (int kb = 0; kb < N_ / 32; ++kb) {
    __syncthreads();                       // protect LDS overwrite
    {                                      // cooperative stage: K 4KB, V^T 4KB
      int kr = tid >> 3, kc = tid & 7;
      *(usv8*)&Klds[kr * 64 + kc * 8] =
          *(const usv8*)(Kh + (size_t)(kb * 32 + kr) * DH_ + kc * 8);
      int vd = tid >> 2, vc = tid & 3;
      *(usv8*)&Vlds[vd * 32 + vc * 8] =
          *(const usv8*)(Vh + (size_t)vd * N_ + kb * 32 + vc * 8);
    }
    if (kb + 1 < N_ / 32)                  // global_prefetch_b8 of next K block
      __builtin_prefetch(Kh + (size_t)((kb + 1) * 32 + (tid >> 3)) * DH_, 0, 1);
    __syncthreads();

    // S = Q K^T : two 16x16 key sub-tiles
    v8f s0 = {}, s1 = {};
#pragma unroll
    for (int c = 0; c < 2; ++c) {
      v16bf b0 = bfcast(cat8(*(const usv8*)&Klds[ln * 64 + c * 32 + off0], *(const usv8*)&Klds[ln * 64 + c * 32 + off0 + 16]));
      s0 = __builtin_amdgcn_wmma_f32_16x16x32_bf16(false, aQ[c], false, b0,
                                                   (short)0, s0, false, false);
      v16bf b1 = bfcast(cat8(*(const usv8*)&Klds[(16 + ln) * 64 + c * 32 + off0], *(const usv8*)&Klds[(16 + ln) * 64 + c * 32 + off0 + 16]));
      s1 = __builtin_amdgcn_wmma_f32_16x16x32_bf16(false, aQ[c], false, b1,
                                                   (short)0, s1, false, false);
    }

    // ---- online softmax (row stats live in each 16-lane half, idx r) ----
    float t[8];
#pragma unroll
    for (int r = 0; r < 8; ++r) t[r] = fmaxf(s0[r], s1[r]);
#pragma unroll
    for (int m = 1; m < 16; m <<= 1)
#pragma unroll
      for (int r = 0; r < 8; ++r) t[r] = fmaxf(t[r], __shfl_xor(t[r], m, 32));
    float rs[8], sc[8];
#pragma unroll
    for (int r = 0; r < 8; ++r) {
      float mn = fmaxf(mrow[r], t[r]);
      sc[r] = __expf(mrow[r] - mn);
      mrow[r] = mn;
      float p0 = __expf(s0[r] - mn);
      float p1 = __expf(s1[r] - mn);
      s0[r] = p0; s1[r] = p1;
      rs[r] = p0 + p1;
    }
#pragma unroll
    for (int m = 1; m < 16; m <<= 1)
#pragma unroll
      for (int r = 0; r < 8; ++r) rs[r] += __shfl_xor(rs[r], m, 32);
#pragma unroll
    for (int r = 0; r < 8; ++r) lrow[r] = lrow[r] * sc[r] + rs[r];
#pragma unroll
    for (int tt = 0; tt < 4; ++tt)
#pragma unroll
      for (int r = 0; r < 8; ++r) o[tt][r] *= sc[r];

    // ---- C-layout P -> A-layout via per-wave LDS bounce ----
    unsigned short* P = Plds[wave];
#pragma unroll
    for (int r = 0; r < 8; ++r) {
      P[(prow + r) * 32 + ln]      = f2bf(s0[r]);
      P[(prow + r) * 32 + 16 + ln] = f2bf(s1[r]);
    }
    asm volatile("s_wait_dscnt 0" ::: "memory");   // wave-internal LDS RAW
    usv8 plo = *(const usv8*)&P[ln * 32 + off0];
    usv8 phi = *(const usv8*)&P[ln * 32 + off0 + 16];
    v16bf aP = bfcast(cat8(plo, phi));

    // O += P @ V  (V^T staged: B[k][dh], lane col dh, two 16B reads at off0, off0+16)
#pragma unroll
    for (int tt = 0; tt < 4; ++tt) {
      v16bf bV = bfcast(cat8(*(const usv8*)&Vlds[(tt * 16 + ln) * 32 + off0], *(const usv8*)&Vlds[(tt * 16 + ln) * 32 + off0 + 16]));
      o[tt] = __builtin_amdgcn_wmma_f32_16x16x32_bf16(false, aP, false, bV,
                                                      (short)0, o[tt], false, false);
    }
  }

  float inv[8];
#pragma unroll
  for (int r = 0; r < 8; ++r) inv[r] = 1.0f / lrow[r];
  float* ob = out + (size_t)(bi * N_) * D_ + (size_t)h * DH_;
#pragma unroll
  for (int tt = 0; tt < 4; ++tt)
#pragma unroll
    for (int r = 0; r < 8; ++r)
      ob[(size_t)(q0 + prow + r) * D_ + tt * 16 + ln] = o[tt][r] * inv[r];
}

// --------------------------------------------------------------------- launch
extern "C" void kernel_launch(void* const* d_in, const int* in_sizes, int n_in,
                              void* d_out, int out_size, void* d_ws, size_t ws_size,
                              hipStream_t stream) {
  (void)in_sizes; (void)n_in; (void)out_size; (void)ws_size;
  const float* z = (const float*)d_in[0];
  const float* W = (const float*)d_in[1];
  float* out = (float*)d_out;

  unsigned short* ws = (unsigned short*)d_ws;
  const size_t headElems = (size_t)B_ * H_ * N_ * DH_;   // 6,291,456
  unsigned short* Qb = ws;
  unsigned short* Kb = Qb + headElems;
  unsigned short* Vt = Kb + headElems;
  unsigned short* zb = Vt + headElems;
  unsigned short* Wb = zb + (size_t)B_ * N_ * D_;

  const int nz8 = B_ * N_ * D_ / 8;       // 786432
  const int nw8 = 3 * D_ * D_ / 8;        // 221184
  cvt_bf16<<<(nz8 + 255) / 256, 256, 0, stream>>>(z, zb, nz8);
  cvt_bf16<<<(nw8 + 255) / 256, 256, 0, stream>>>(W, Wb, nw8);

  // 256 row-tiles (32 rows) * 36 col-tiles, 8 wave-tiles per block
  qkv_gemm<<<(256 * 36) / 8, 256, 0, stream>>>(zb, Wb, Qb, Kb, Vt);
  // 48 heads * 16 q-blocks of 128 rows
  attn<<<B_ * H_ * (N_ / 128), 256, 0, stream>>>(Qb, Kb, Vt, out);
}
